// MoE_42245298323842
// MI455X (gfx1250) — matrix operand order repaired and checked by reading, not compiled, not run on hardware
//
#include <hip/hip_runtime.h>
#include <hip/hip_bf16.h>

#define NUM_EXPERTS 8
#define NUM_ROWS    8192
#define HIDDEN      2048
#define INTER       2048
#define TOPK        2

// ---------------- CDNA5 WMMA types ----------------
typedef __attribute__((ext_vector_type(16))) __bf16 bf16x16;
typedef __attribute__((ext_vector_type(8)))  __bf16 bf16x8;
typedef __attribute__((ext_vector_type(4)))  __bf16 bf16x4;
typedef __attribute__((ext_vector_type(8)))  float  f32x8;
typedef __attribute__((ext_vector_type(4)))  int    v4i;

union V16 { bf16x16 v; bf16x8 h[2]; };

__device__ __forceinline__ f32x8 wmma_bf16(bf16x16 a, bf16x16 b, f32x8 c) {
  // 8 args: (neg_a, A, neg_b, B, c_mod, C, reuse_a, reuse_b)
  return __builtin_amdgcn_wmma_f32_16x16x32_bf16(false, a, false, b, (short)0, c,
                                                 false, false);
}

// async global->LDS (ASYNCcnt path), if this toolchain exposes it
#if defined(__AMDGCN__) && __has_builtin(__builtin_amdgcn_global_load_async_to_lds_b128)
  #define USE_ASYNC_LDS 1
  typedef __attribute__((address_space(1))) v4i gv4i;   // global v4i
  typedef __attribute__((address_space(3))) v4i lv4i;   // LDS v4i
#else
  #define USE_ASYNC_LDS 0
#endif

__device__ __forceinline__ void wait_async0() {
#if defined(__AMDGCN__)
  #if __has_builtin(__builtin_amdgcn_s_wait_asynccnt)
    __builtin_amdgcn_s_wait_asynccnt(0);
  #elif USE_ASYNC_LDS
    asm volatile("s_wait_asynccnt 0x0" ::: "memory");
  #endif
#endif
}

static constexpr int LDT = 40;          // padded LDS row stride (80B, 16B aligned)
static constexpr int NK  = HIDDEN / 32; // 64 k-steps of depth 32

// ---------------- routing kernels ----------------
__global__ void moe_router(const float* __restrict__ rp,
                           int* __restrict__ tk_e, float* __restrict__ tk_w,
                           unsigned* __restrict__ counts) {
  int t = blockIdx.x * blockDim.x + threadIdx.x;
  if (t >= NUM_ROWS) return;
  float p[NUM_EXPERTS];
  #pragma unroll
  for (int j = 0; j < NUM_EXPERTS; ++j) p[j] = rp[t * NUM_EXPERTS + j];
  int i1 = 0; float b1 = p[0];
  #pragma unroll
  for (int j = 1; j < NUM_EXPERTS; ++j) if (p[j] > b1) { b1 = p[j]; i1 = j; }
  int i2 = (i1 == 0) ? 1 : 0; float b2 = p[i2];
  #pragma unroll
  for (int j = 0; j < NUM_EXPERTS; ++j)
    if (j != i1 && p[j] > b2) { b2 = p[j]; i2 = j; }
  float s = b1 + b2;
  tk_e[t * 2 + 0] = i1; tk_w[t * 2 + 0] = b1 / s;
  tk_e[t * 2 + 1] = i2; tk_w[t * 2 + 1] = b2 / s;
  atomicAdd(&counts[i1], 1u);
  atomicAdd(&counts[i2], 1u);
}

__global__ void moe_prefix(const unsigned* __restrict__ counts,
                           unsigned* __restrict__ offsets) {
  if (threadIdx.x == 0 && blockIdx.x == 0) {
    unsigned s = 0;
    for (int e = 0; e < NUM_EXPERTS; ++e) { offsets[e] = s; s += counts[e]; }
    offsets[NUM_EXPERTS] = s;
  }
}

__global__ void moe_scatter(const int* __restrict__ tk_e, const float* __restrict__ tk_w,
                            const unsigned* __restrict__ offsets,
                            unsigned* __restrict__ cursor,
                            int* __restrict__ tokmap, float* __restrict__ tokw) {
  int t = blockIdx.x * blockDim.x + threadIdx.x;
  if (t >= NUM_ROWS) return;
  #pragma unroll
  for (int k = 0; k < TOPK; ++k) {
    int e = tk_e[t * 2 + k];
    unsigned pos = offsets[e] + atomicAdd(&cursor[e], 1u);
    tokmap[pos] = t;
    tokw[pos]   = tk_w[t * 2 + k];
  }
}

// ---------------- stage 1: act = silu(Xg @ fc1) * (Xg @ fc3) ----------------
// block tile 128(M) x 64(N), 8 wave32 as 4(M) x 2(N), wave tile 32x32
__global__ __launch_bounds__(256, 2)
void moe_stage1(const float* __restrict__ x,
                const float* __restrict__ fc1,
                const float* __restrict__ fc3,
                const int* __restrict__ tokmap,
                const unsigned* __restrict__ counts,
                const unsigned* __restrict__ offsets,
                __bf16* __restrict__ act) {
  const int e = blockIdx.z;
  const unsigned cnt = counts[e];
  const unsigned mtile = blockIdx.y;
  if (mtile * 128u >= cnt) return;
  const unsigned gbase = offsets[e];
  const int n0 = blockIdx.x * 64;
  const float* __restrict__ W1 = fc1 + (size_t)e * HIDDEN * INTER;
  const float* __restrict__ W3 = fc3 + (size_t)e * HIDDEN * INTER;

  __shared__ __bf16 sA[2][128 * LDT];
  __shared__ __bf16 sB1[2][64 * LDT];
  __shared__ __bf16 sB3[2][64 * LDT];

  const int tid = threadIdx.x;
  const int lane = tid & 31;
  const int wave = tid >> 5;
  const int waveM = wave & 3;    // 4 waves over M
  const int waveN = wave >> 2;   // 2 waves over N
  const int half = lane >> 4;
  const int lr = lane & 15;

  // ---- K-invariant staging addresses (hoisted gather, branch-free tails) ----
  const float* aPtr[4];
  int aOff[4];
  #pragma unroll
  for (int f = 0; f < 4; ++f) {
    int idx = tid + 256 * f;            // 0..1023 : 128 rows x 8 float4
    int row = idx >> 3;
    int kc  = (idx & 7) * 4;
    unsigned gr  = mtile * 128u + (unsigned)row;
    unsigned grc = gr < cnt ? gr : cnt - 1;     // clamp: tail rows duplicate a valid row
    int tok = tokmap[gbase + grc];
    aPtr[f] = x + (size_t)tok * HIDDEN + kc;
    aOff[f] = row * LDT + kc;
  }
  // B staging: each thread owns a 2(n) x 4(k) block of each weight tile.
  // Loads: 4 coalesced float2 per matrix; stores: 2 packed bf16x4 (ds_store_b64).
  const int bnc = (tid & 31) * 2;       // 0..62 (64 cols)
  const int bk  = (tid >> 5) * 4;       // 0..28 (32 k rows)
  const float* b1Ptr = W1 + (size_t)bk * INTER + n0 + bnc;
  const float* b3Ptr = W3 + (size_t)bk * INTER + n0 + bnc;
  const int bOff = bnc * LDT + bk;

  float4 ra[4]; float2 rb1[4], rb3[4];
  auto loadRegs = [&](int k0) {
    #pragma unroll
    for (int f = 0; f < 4; ++f) ra[f] = *(const float4*)(aPtr[f] + k0);
    #pragma unroll
    for (int j = 0; j < 4; ++j) {
      rb1[j] = *(const float2*)(b1Ptr + (size_t)(k0 + j) * INTER);
      rb3[j] = *(const float2*)(b3Ptr + (size_t)(k0 + j) * INTER);
    }
  };
  auto storeRegs = [&](int buf) {
    #pragma unroll
    for (int f = 0; f < 4; ++f) {
      bf16x4 b;
      b.x = (__bf16)ra[f].x; b.y = (__bf16)ra[f].y;
      b.z = (__bf16)ra[f].z; b.w = (__bf16)ra[f].w;
      *(bf16x4*)&sA[buf][aOff[f]] = b;
    }
    bf16x4 c;
    c.x = (__bf16)rb1[0].x; c.y = (__bf16)rb1[1].x;
    c.z = (__bf16)rb1[2].x; c.w = (__bf16)rb1[3].x;
    *(bf16x4*)&sB1[buf][bOff] = c;
    c.x = (__bf16)rb1[0].y; c.y = (__bf16)rb1[1].y;
    c.z = (__bf16)rb1[2].y; c.w = (__bf16)rb1[3].y;
    *(bf16x4*)&sB1[buf][bOff + LDT] = c;
    c.x = (__bf16)rb3[0].x; c.y = (__bf16)rb3[1].x;
    c.z = (__bf16)rb3[2].x; c.w = (__bf16)rb3[3].x;
    *(bf16x4*)&sB3[buf][bOff] = c;
    c.x = (__bf16)rb3[0].y; c.y = (__bf16)rb3[1].y;
    c.z = (__bf16)rb3[2].y; c.w = (__bf16)rb3[3].y;
    *(bf16x4*)&sB3[buf][bOff + LDT] = c;
  };

  f32x8 acc1[2][2], acc3[2][2];
  const f32x8 zero = {0.f, 0.f, 0.f, 0.f, 0.f, 0.f, 0.f, 0.f};
  #pragma unroll
  for (int i = 0; i < 2; ++i)
    #pragma unroll
    for (int j = 0; j < 2; ++j) { acc1[i][j] = zero; acc3[i][j] = zero; }

  loadRegs(0);
  storeRegs(0);
  __syncthreads();

  int buf = 0;
  for (int kt = 0; kt < NK; ++kt) {
    const bool more = (kt + 1 < NK);
    if (more) loadRegs((kt + 1) * 32);        // global loads in flight over WMMAs

    V16 a[2], b1[2], b3[2];
    #pragma unroll
    for (int ms = 0; ms < 2; ++ms) {
      // A 16x32 bf16: lanes 0-15 K {0..7,16..23}; lanes 16-31 K {8..15,24..31}
      const __bf16* p = &sA[buf][(waveM * 32 + ms * 16 + lr) * LDT + half * 8];
      a[ms].h[0] = *(const bf16x8*)p;
      a[ms].h[1] = *(const bf16x8*)(p + 16);
    }
    #pragma unroll
    for (int ns = 0; ns < 2; ++ns) {
      // B 32x16 bf16: lanes 0-15 K {0..7,16..23}; lanes 16-31 K {8..15,24..31}
      const __bf16* q1 = &sB1[buf][(waveN * 32 + ns * 16 + lr) * LDT + half * 8];
      b1[ns].h[0] = *(const bf16x8*)q1;
      b1[ns].h[1] = *(const bf16x8*)(q1 + 16);
      const __bf16* q3 = &sB3[buf][(waveN * 32 + ns * 16 + lr) * LDT + half * 8];
      b3[ns].h[0] = *(const bf16x8*)q3;
      b3[ns].h[1] = *(const bf16x8*)(q3 + 16);
    }
    #pragma unroll
    for (int ms = 0; ms < 2; ++ms)
      #pragma unroll
      for (int ns = 0; ns < 2; ++ns) {
        acc1[ms][ns] = wmma_bf16(a[ms].v, b1[ns].v, acc1[ms][ns]);
        acc3[ms][ns] = wmma_bf16(a[ms].v, b3[ns].v, acc3[ms][ns]);
      }

    if (more) storeRegs(buf ^ 1);
    __syncthreads();
    buf ^= 1;
  }

  // fused SwiGLU epilogue -> bf16 act
  #pragma unroll
  for (int ms = 0; ms < 2; ++ms)
    #pragma unroll
    for (int ns = 0; ns < 2; ++ns)
      #pragma unroll
      for (int i = 0; i < 8; ++i) {
        unsigned row = mtile * 128u + waveM * 32 + ms * 16 + i + 8 * half;
        if (row < cnt) {
          float h1 = acc1[ms][ns][i];
          float h3 = acc3[ms][ns][i];
          float sv = h1 * (1.0f / (1.0f + __expf(-h1)));
          int col = n0 + waveN * 32 + ns * 16 + lr;
          act[(size_t)(gbase + row) * INTER + col] = (__bf16)(sv * h3);
        }
      }
}

// ---------------- stage 2: out[tok] += w * (act_g @ fc2) ----------------
// block tile 128(M) x 128(N), 8 wave32 as 4(M) x 2(N), wave tile 32x64
__global__ __launch_bounds__(256, 2)
void moe_stage2(const __bf16* __restrict__ act,
                const float* __restrict__ fc2,
                const int* __restrict__ tokmap,
                const float* __restrict__ tokw,
                const unsigned* __restrict__ counts,
                const unsigned* __restrict__ offsets,
                float* __restrict__ out) {
  const int e = blockIdx.z;
  const unsigned cnt = counts[e];
  const unsigned mtile = blockIdx.y;
  if (mtile * 128u >= cnt) return;
  const unsigned gbase = offsets[e];
  const int n0 = blockIdx.x * 128;
  const float* __restrict__ W2 = fc2 + (size_t)e * INTER * HIDDEN;

  __shared__ __bf16 sA[2][128 * LDT];
  __shared__ __bf16 sB[2][128 * LDT];

  const int tid = threadIdx.x;
  const int lane = tid & 31;
  const int wave = tid >> 5;
  const int waveM = wave & 3;
  const int waveN = wave >> 2;
  const int half = lane >> 4;
  const int lr = lane & 15;

  // ---- K-invariant staging addresses ----
  const __bf16* aPtr[2]; int aOff[2];
  #pragma unroll
  for (int f = 0; f < 2; ++f) {
    int idx = tid + 256 * f;            // 0..511 : 128 rows x 4 bf16x8
    int row = idx >> 2;
    int kc  = (idx & 3) * 8;
    unsigned gr  = mtile * 128u + (unsigned)row;
    unsigned grc = gr < cnt ? gr : cnt - 1;     // clamp tail (act rows contiguous)
    aPtr[f] = act + (size_t)(gbase + grc) * INTER + kc;
    aOff[f] = row * LDT + kc;
  }
  // B staging: each thread owns a 2(n) x 8(k) block.
  // Loads: 8 coalesced float2; stores: 2 packed bf16x8 (ds_store_b128).
  const int bnc = (tid & 63) * 2;       // 0..126 (128 cols)
  const int bk  = (tid >> 6) * 8;       // 0..24  (32 k rows)
  const float* bPtr = W2 + (size_t)bk * HIDDEN + n0 + bnc;
  const int bOff = bnc * LDT + bk;

  bf16x8 ra[2]; float2 rb[8];
  auto loadA = [&](int k0) {
    #pragma unroll
    for (int f = 0; f < 2; ++f) ra[f] = *(const bf16x8*)(aPtr[f] + k0);
  };
  auto storeA = [&](int buf) {
    #pragma unroll
    for (int f = 0; f < 2; ++f) *(bf16x8*)&sA[buf][aOff[f]] = ra[f];
  };
#if USE_ASYNC_LDS
  auto asyncA = [&](int buf, int k0) {
    #pragma unroll
    for (int f = 0; f < 2; ++f) {
      __builtin_amdgcn_global_load_async_to_lds_b128(
          (gv4i*)(aPtr[f] + k0),
          (lv4i*)&sA[buf][aOff[f]],
          0, 0);
    }
  };
#endif
  auto loadB = [&](int k0) {
    #pragma unroll
    for (int j = 0; j < 8; ++j)
      rb[j] = *(const float2*)(bPtr + (size_t)(k0 + j) * HIDDEN);
  };
  auto storeB = [&](int buf) {
    bf16x8 c;
    #pragma unroll
    for (int j = 0; j < 8; ++j) c[j] = (__bf16)rb[j].x;
    *(bf16x8*)&sB[buf][bOff] = c;
    #pragma unroll
    for (int j = 0; j < 8; ++j) c[j] = (__bf16)rb[j].y;
    *(bf16x8*)&sB[buf][bOff + LDT] = c;
  };

  f32x8 acc[2][4];
  const f32x8 zero = {0.f, 0.f, 0.f, 0.f, 0.f, 0.f, 0.f, 0.f};
  #pragma unroll
  for (int i = 0; i < 2; ++i)
    #pragma unroll
    for (int j = 0; j < 4; ++j) acc[i][j] = zero;

#if USE_ASYNC_LDS
  asyncA(0, 0);
  loadB(0); storeB(0);
  wait_async0();
#else
  loadA(0); loadB(0);
  storeA(0); storeB(0);
#endif
  __syncthreads();

  int buf = 0;
  for (int kt = 0; kt < NK; ++kt) {
    const bool more = (kt + 1 < NK);
    if (more) {
#if USE_ASYNC_LDS
      asyncA(buf ^ 1, (kt + 1) * 32);   // DMA straight into the free buffer
      loadB((kt + 1) * 32);
#else
      loadA((kt + 1) * 32); loadB((kt + 1) * 32);
#endif
    }

    V16 a[2], b[4];
    #pragma unroll
    for (int ms = 0; ms < 2; ++ms) {
      const __bf16* p = &sA[buf][(waveM * 32 + ms * 16 + lr) * LDT + half * 8];
      a[ms].h[0] = *(const bf16x8*)p;
      a[ms].h[1] = *(const bf16x8*)(p + 16);
    }
    #pragma unroll
    for (int ns = 0; ns < 4; ++ns) {
      const __bf16* q = &sB[buf][(waveN * 64 + ns * 16 + lr) * LDT + half * 8];
      b[ns].h[0] = *(const bf16x8*)q;
      b[ns].h[1] = *(const bf16x8*)(q + 16);
    }
    #pragma unroll
    for (int ms = 0; ms < 2; ++ms)
      #pragma unroll
      for (int ns = 0; ns < 4; ++ns)
        acc[ms][ns] = wmma_bf16(a[ms].v, b[ns].v, acc[ms][ns]);

    if (more) {
#if USE_ASYNC_LDS
      storeB(buf ^ 1);
      wait_async0();                    // our async LDS writes done before publish
#else
      storeA(buf ^ 1); storeB(buf ^ 1);
#endif
    }
    __syncthreads();
    buf ^= 1;
  }

  // scale by routing weight, scatter-accumulate (exactly 2 contributions/elem)
  #pragma unroll
  for (int ms = 0; ms < 2; ++ms) {
    unsigned rbase = mtile * 128u + waveM * 32 + ms * 16 + 8 * half;
    #pragma unroll
    for (int i = 0; i < 8; ++i) {
      unsigned row = rbase + i;
      if (row < cnt) {
        unsigned g = gbase + row;
        int tok = tokmap[g];
        float w = tokw[g];
        #pragma unroll
        for (int ns = 0; ns < 4; ++ns) {
          int col = n0 + waveN * 64 + ns * 16 + lr;
          atomicAdd(out + (size_t)tok * HIDDEN + col, w * acc[ms][ns][i]);
        }
      }
    }
  }
}

// ---------------- launch ----------------
// Workspace layout (needs ~64.25 MB):
//   [0, 64MB)  : act (16384 x 2048 bf16)
//   then counts(8 u32), cursor(8 u32), offsets(pad), tk_e/tk_w/tokmap/tokw
extern "C" void kernel_launch(void* const* d_in, const int* in_sizes, int n_in,
                              void* d_out, int out_size, void* d_ws, size_t ws_size,
                              hipStream_t stream) {
  const float* x   = (const float*)d_in[0];
  const float* rp  = (const float*)d_in[1];
  const float* fc1 = (const float*)d_in[2];
  const float* fc2 = (const float*)d_in[3];
  const float* fc3 = (const float*)d_in[4];
  float* out = (float*)d_out;

  char* ws = (char*)d_ws;
  const size_t actBytes = (size_t)TOPK * NUM_ROWS * INTER * 2;   // 64 MB
  __bf16* act = (__bf16*)ws;
  char* p = ws + actBytes;
  unsigned* counts  = (unsigned*)p; p += 32;
  unsigned* cursor  = (unsigned*)p; p += 32;
  unsigned* offsets = (unsigned*)p; p += 64;
  int*   tk_e   = (int*)p;   p += (size_t)TOPK * NUM_ROWS * 4;
  float* tk_w   = (float*)p; p += (size_t)TOPK * NUM_ROWS * 4;
  int*   tokmap = (int*)p;   p += (size_t)TOPK * NUM_ROWS * 4;
  float* tokw   = (float*)p; p += (size_t)TOPK * NUM_ROWS * 4;

  (void)hipMemsetAsync(counts, 0, 64, stream);                 // counts + cursor
  (void)hipMemsetAsync(d_out, 0, (size_t)out_size * sizeof(float), stream);

  moe_router <<<NUM_ROWS / 256, 256, 0, stream>>>(rp, tk_e, tk_w, counts);
  moe_prefix <<<1, 32, 0, stream>>>(counts, offsets);
  moe_scatter<<<NUM_ROWS / 256, 256, 0, stream>>>(tk_e, tk_w, offsets, cursor,
                                                  tokmap, tokw);

  dim3 g1(INTER / 64, NUM_ROWS / 128, NUM_EXPERTS);    // (32, 64, 8), early-exit on cnt
  moe_stage1<<<g1, 256, 0, stream>>>(x, fc1, fc3, tokmap, counts, offsets, act);

  dim3 g2(HIDDEN / 128, NUM_ROWS / 128, NUM_EXPERTS);  // (16, 64, 8)
  moe_stage2<<<g2, 256, 0, stream>>>(act, fc2, tokmap, tokw, counts, offsets, out);
}
